// Gated_GNNML_55147380080745
// MI455X (gfx1250) — matrix operand order repaired and checked by reading, not compiled, not run on hardware
//
#include <hip/hip_runtime.h>

typedef __attribute__((ext_vector_type(16))) _Float16 v16h;
typedef __attribute__((ext_vector_type(8)))  float    v8f;

__device__ __forceinline__ v8f wmma16(v16h a, v16h b, v8f c) {
  // D = A(16x32,f16) * B(32x16,f16) + C(16x16,f32)
  return __builtin_amdgcn_wmma_f32_16x16x32_f16(
      /*neg_a=*/false, a, /*neg_b=*/false, b,
      /*c_mod=*/(short)0, c, /*reuse_a=*/false, /*reuse_b=*/false);
}

// B fragment: lane l holds column n = nt*16 + l%16 of W[K x Nd]; element e is k in table order.
__device__ __forceinline__ v16h load_bfrag(const _Float16* __restrict__ W,
                                           int Kdim, int Ndim, int nt, int ks, int lane) {
  int n = nt * 16 + (lane & 15);
  int h = lane >> 4;
  v16h z = {};
#pragma unroll
  for (int e = 0; e < 16; ++e) {
    int k = ks * 32 + ((e < 8) ? (h * 8 + e) : (16 + h * 8 + (e - 8)));
    if (k < Kdim) z[e] = W[(size_t)k * Ndim + n];
  }
  return z;
}

// ---------------- weight f32 -> f16 conversion ----------------
struct CvtArgs { const float* src[17]; int n[17]; int off[17]; };

__global__ void cvt_kernel(CvtArgs a, _Float16* __restrict__ dst) {
  int b = blockIdx.x;
  const float* s = a.src[b];
  _Float16* d = dst + a.off[b];
  for (int i = threadIdx.x; i < a.n[b]; i += blockDim.x) d[i] = (_Float16)s[i];
}

__global__ void zero_kernel(float* __restrict__ p, long long n) {
  long long i = (long long)blockIdx.x * blockDim.x + threadIdx.x;
  long long stride = (long long)gridDim.x * blockDim.x;
  for (; i < n; i += stride) p[i] = 0.0f;
}

// ---------------- edge scatter: agg[dst] += feat[src] ----------------
template<int D>
__global__ __launch_bounds__(256)
void scatter_kernel(const float* __restrict__ feat, const int* __restrict__ ei,
                    int E, float* __restrict__ agg) {
  constexpr int CH = D / 4;
  long long t = (long long)blockIdx.x * blockDim.x + threadIdx.x;
  if (t >= (long long)E * CH) return;
  int e = (int)(t / CH);
  int c = (int)(t % CH);
  int s = ei[e];
  int d = ei[(size_t)E + e];
  const float4 v = *(const float4*)(feat + (size_t)s * D + (size_t)c * 4);
  float* o = agg + (size_t)d * D + (size_t)c * 4;
  unsafeAtomicAdd(o + 0, v.x);
  unsafeAtomicAdd(o + 1, v.y);
  unsafeAtomicAdd(o + 2, v.z);
  unsafeAtomicAdd(o + 3, v.w);
}

// ---------------- fused block + gate (WMMA) ----------------
// h = relu(x@Wfc1+b) + relu(agg@Wcv+b) + relu((x@Wfc2+b)*(x@Wfc3+b))   [D -> D]
// FINAL=false: out = sigmoid(h@Wg1+bg1) * (h@Wg2+bg2)                  [D -> D/2]
// FINAL=true : out = h@Wg1 + bg1                                       [16 -> 16]
template<int D, bool FINAL>
__global__ __launch_bounds__(256)
void block_kernel(const float* __restrict__ xin, const float* __restrict__ agg,
                  const _Float16* __restrict__ w_fc1, const float* __restrict__ b_fc1,
                  const _Float16* __restrict__ w_cv,  const float* __restrict__ b_cv,
                  const _Float16* __restrict__ w_fc2, const float* __restrict__ b_fc2,
                  const _Float16* __restrict__ w_fc3, const float* __restrict__ b_fc3,
                  const _Float16* __restrict__ w_g1,  const float* __restrict__ b_g1,
                  const _Float16* __restrict__ w_g2,  const float* __restrict__ b_g2,
                  float* __restrict__ out, int N) {
  constexpr int KS  = (D + 31) / 32;      // k-steps of 32
  constexpr int NT  = D / 16;             // output column tiles of h
  constexpr int DG  = FINAL ? 16 : D / 2; // gate/final output dim
  constexpr int NTG = DG / 16;

  __shared__ float lds_all[8 * 16 * D];

  const int wave  = threadIdx.x >> 5;
  const int lane  = threadIdx.x & 31;
  const int tiles = (N + 15) >> 4;
  const int tile  = blockIdx.x * 8 + wave;
  if (tile >= tiles) return;

  float* lds = lds_all + wave * 16 * D;
  const int m    = lane & 15;   // A: row within tile; C: column within tile
  const int kh   = lane >> 4;   // A: k-half; C: row-half
  const int row0 = tile * 16;

  // ---- A fragments of x and agg (f32 -> f16, hardware A layout) ----
  v16h ax[KS], ag[KS];
  {
    int row = row0 + m; if (row >= N) row = N - 1;
    const float* px = xin + (size_t)row * D;
    const float* pa = agg + (size_t)row * D;
#pragma unroll
    for (int ks = 0; ks < KS; ++ks) {
      v16h a = {}, g = {};
#pragma unroll
      for (int e = 0; e < 16; ++e) {
        int k = ks * 32 + ((e < 8) ? (kh * 8 + e) : (16 + kh * 8 + (e - 8)));
        if (k < D) { a[e] = (_Float16)px[k]; g[e] = (_Float16)pa[k]; }
      }
      ax[ks] = a; ag[ks] = g;
    }
  }

  // ---- h = relu(fc1) + relu(conv) + relu(fc2*fc3), stored to LDS ----
#pragma unroll
  for (int nt = 0; nt < NT; ++nt) {
    v8f c1 = {}, cc = {}, c2 = {}, c3 = {};
#pragma unroll
    for (int ks = 0; ks < KS; ++ks) {
      c1 = wmma16(ax[ks], load_bfrag(w_fc1, D, D, nt, ks, lane), c1);
      cc = wmma16(ag[ks], load_bfrag(w_cv,  D, D, nt, ks, lane), cc);
      c2 = wmma16(ax[ks], load_bfrag(w_fc2, D, D, nt, ks, lane), c2);
      c3 = wmma16(ax[ks], load_bfrag(w_fc3, D, D, nt, ks, lane), c3);
    }
    const int col = nt * 16 + m;
    const float bb1 = b_fc1[col], bbc = b_cv[col], bb2 = b_fc2[col], bb3 = b_fc3[col];
#pragma unroll
    for (int r = 0; r < 8; ++r) {
      float h = fmaxf(c1[r] + bb1, 0.0f)
              + fmaxf(cc[r] + bbc, 0.0f)
              + fmaxf((c2[r] + bb2) * (c3[r] + bb3), 0.0f);
      lds[(r + 8 * kh) * D + col] = h;  // same-wave LDS: DS ops are in-order
    }
  }

  // ---- reload h as A fragments (C layout -> A layout via LDS) ----
  v16h ah[KS];
  {
    const float* hrow = lds + m * D;
#pragma unroll
    for (int ks = 0; ks < KS; ++ks) {
      v16h a = {};
#pragma unroll
      for (int e = 0; e < 16; ++e) {
        int k = ks * 32 + ((e < 8) ? (kh * 8 + e) : (16 + kh * 8 + (e - 8)));
        if (k < D) a[e] = (_Float16)hrow[k];
      }
      ah[ks] = a;
    }
  }

  // ---- gate / final linear ----
#pragma unroll
  for (int nt = 0; nt < NTG; ++nt) {
    v8f u1 = {}, u2 = {};
#pragma unroll
    for (int ks = 0; ks < KS; ++ks) {
      u1 = wmma16(ah[ks], load_bfrag(w_g1, D, DG, nt, ks, lane), u1);
      if (!FINAL)
        u2 = wmma16(ah[ks], load_bfrag(w_g2, D, DG, nt, ks, lane), u2);
    }
    const int col = nt * 16 + m;
    const float bg1 = b_g1[col];
    const float bg2 = FINAL ? 0.0f : b_g2[col];
#pragma unroll
    for (int r = 0; r < 8; ++r) {
      int grow = row0 + r + 8 * kh;
      if (grow < N) {
        float v;
        if (FINAL) v = u1[r] + bg1;
        else       v = (u2[r] + bg2) / (1.0f + __expf(-(u1[r] + bg1)));
        out[(size_t)grow * DG + col] = v;
      }
    }
  }
}

// ---------------- host orchestration ----------------
extern "C" void kernel_launch(void* const* d_in, const int* in_sizes, int n_in,
                              void* d_out, int out_size, void* d_ws, size_t ws_size,
                              hipStream_t stream) {
  (void)n_in; (void)out_size; (void)ws_size;
  const float* x = (const float*)d_in[0];
  const int N = in_sizes[0] / 64;
  const int* ei = (const int*)d_in[35];          // edge_index [2,E] int32
  const int E = in_sizes[35] / 2;
  auto W = [&](int i) { return (const float*)d_in[i]; };

  // workspace layout
  char* ws = (char*)d_ws;
  _Float16* wh = (_Float16*)ws;                  // 26880 f16 weights
  size_t off = 65536;
  float* agg1 = (float*)(ws + off); off += (size_t)N * 64 * 4;
  float* agg2 = (float*)(ws + off); off += (size_t)N * 32 * 4;
  float* agg3 = (float*)(ws + off); off += (size_t)N * 16 * 4;
  float* g1   = (float*)(ws + off); off += (size_t)N * 32 * 4;
  float* g2   = (float*)(ws + off); off += (size_t)N * 16 * 4;

  // convert all 17 weight matrices to f16 (packed, 32B-aligned rows)
  static const int srcs[17]  = {3,1,5,7, 9,11, 15,13,17,19, 21,23, 27,25,29,31, 33};
  static const int sizes[17] = {4096,4096,4096,4096, 2048,2048,
                                1024,1024,1024,1024, 512,512,
                                256,256,256,256, 256};
  CvtArgs ca; int o = 0;
  for (int i = 0; i < 17; ++i) { ca.src[i] = W(srcs[i]); ca.n[i] = sizes[i]; ca.off[i] = o; o += sizes[i]; }
  cvt_kernel<<<17, 256, 0, stream>>>(ca, wh);

  // zero agg1..agg3 (contiguous)
  zero_kernel<<<2048, 256, 0, stream>>>(agg1, (long long)N * (64 + 32 + 16));

  const _Float16 *fc11 = wh + 0,     *cv11 = wh + 4096,  *fc12 = wh + 8192,  *fc13 = wh + 12288,
                 *g1f1 = wh + 16384, *g1f2 = wh + 18432,
                 *fc21 = wh + 20480, *cv21 = wh + 21504, *fc22 = wh + 22528, *fc23 = wh + 23552,
                 *g2f1 = wh + 24576, *g2f2 = wh + 25088,
                 *fc31 = wh + 25600, *cv31 = wh + 25856, *fc32 = wh + 26112, *fc33 = wh + 26368,
                 *fc2w = wh + 26624;

  const int tiles = (N + 15) / 16;
  const int tblocks = (tiles + 7) / 8;

  { long long tot = (long long)E * 16;
    scatter_kernel<64><<<(int)((tot + 255) / 256), 256, 0, stream>>>(x, ei, E, agg1); }
  block_kernel<64, false><<<tblocks, 256, 0, stream>>>(
      x, agg1, fc11, W(4), cv11, W(2), fc12, W(6), fc13, W(8),
      g1f1, W(10), g1f2, W(12), g1, N);

  { long long tot = (long long)E * 8;
    scatter_kernel<32><<<(int)((tot + 255) / 256), 256, 0, stream>>>(g1, ei, E, agg2); }
  block_kernel<32, false><<<tblocks, 256, 0, stream>>>(
      g1, agg2, fc21, W(16), cv21, W(14), fc22, W(18), fc23, W(20),
      g2f1, W(22), g2f2, W(24), g2, N);

  { long long tot = (long long)E * 4;
    scatter_kernel<16><<<(int)((tot + 255) / 256), 256, 0, stream>>>(g2, ei, E, agg3); }
  block_kernel<16, true><<<tblocks, 256, 0, stream>>>(
      g2, agg3, fc31, W(28), cv31, W(26), fc32, W(30), fc33, W(32),
      fc2w, W(34), fc2w, W(34), (float*)d_out, N);
}
